// MultiHeadAttention_RoPE_11038065951105
// MI455X (gfx1250) — matrix operand order repaired and checked by reading, not compiled, not run on hardware
//
#include <hip/hip_runtime.h>
#include <hip/hip_bf16.h>

typedef __attribute__((ext_vector_type(16))) _Float16 v16h;
typedef __attribute__((ext_vector_type(8)))  _Float16 v8h;
typedef __attribute__((ext_vector_type(2)))  _Float16 v2h;
typedef __attribute__((ext_vector_type(8)))  float    v8f;

// ---------------------------------------------------------------------------
// helpers
// ---------------------------------------------------------------------------
__device__ inline v16h mk16(v8h lo, v8h hi) {
    v16h r;
#pragma unroll
    for (int j = 0; j < 8; ++j) { r[j] = lo[j]; r[8 + j] = hi[j]; }
    return r;
}

__device__ inline v8f wmma_f16(v16h a, v16h b, v8f c) {
    // D = A(16x32 f16) * B(32x16 f16) + C(16x16 f32)
    return __builtin_amdgcn_wmma_f32_16x16x32_f16(
        /*neg_a=*/false, a, /*neg_b=*/false, b,
        /*c_mod=*/(short)0, c, /*reuse_a=*/false, /*reuse_b=*/false);
}

__device__ inline void lds_fence_wait() {
    __builtin_amdgcn_wave_barrier();
    asm volatile("s_wait_dscnt 0" ::: "memory");
    __builtin_amdgcn_wave_barrier();
}

// ---------------------------------------------------------------------------
// f32 -> f16 vectorized convert (8 elements / thread)
// ---------------------------------------------------------------------------
__global__ __launch_bounds__(256) void cvt_f16(
    const float* __restrict__ x, _Float16* __restrict__ y, int total)
{
    const int i = (blockIdx.x * blockDim.x + threadIdx.x) * 8;
    if (i >= total) return;
    v8h r;
#pragma unroll
    for (int j = 0; j < 8; ++j) r[j] = (_Float16)x[i + j];
    *(v8h*)(y + i) = r;
}

// ---------------------------------------------------------------------------
// All-f16 GEMM with bias:  out[M,N] = A[M,K] @ Wh[N,K]^T + bias[N]
// Register-blocked: each wave computes a 32x64 tile (2 A-frags x 4 B-frags
// -> 8 WMMAs per K-chunk). Block = 8 waves as 4(M) x 2(N) -> 128x128 tile.
// ---------------------------------------------------------------------------
template <bool OUT_F16>
__global__ __launch_bounds__(256) void gemm_bias_wmma(
    const _Float16* __restrict__ A, const _Float16* __restrict__ Wh,
    const float* __restrict__ bias, void* __restrict__ out_,
    int M, int N, int K)
{
    const int lane = threadIdx.x & 31;
    const int wid  = threadIdx.x >> 5;
    const int half = lane >> 4;
    const int l15  = lane & 15;
    const int m0   = blockIdx.y * 128 + (wid >> 1) * 32;
    const int n0   = blockIdx.x * 128 + (wid & 1) * 64;

    const _Float16* ap0 = A + (size_t)(m0 + l15) * K;        // A-frag rows m0..m0+15
    const _Float16* ap1 = A + (size_t)(m0 + 16 + l15) * K;   // A-frag rows m0+16..m0+31
    const _Float16* wp[4];
#pragma unroll
    for (int t = 0; t < 4; ++t)
        wp[t] = Wh + (size_t)(n0 + t * 16 + l15) * K + half * 8;

    v8f c[2][4] = {};
    for (int kk = 0; kk < K; kk += 32) {
        v16h a0 = mk16(*(const v8h*)(ap0 + kk + half * 8),
                       *(const v8h*)(ap0 + kk + 16 + half * 8));
        v16h a1 = mk16(*(const v8h*)(ap1 + kk + half * 8),
                       *(const v8h*)(ap1 + kk + 16 + half * 8));
#pragma unroll
        for (int t = 0; t < 4; ++t) {
            v16h b = mk16(*(const v8h*)(wp[t] + kk),
                          *(const v8h*)(wp[t] + kk + 16));
            c[0][t] = wmma_f16(a0, b, c[0][t]);
            c[1][t] = wmma_f16(a1, b, c[1][t]);
        }
    }

#pragma unroll
    for (int t = 0; t < 4; ++t) {
        const int nB = n0 + t * 16 + l15;
        const float bvl = bias[nB];
#pragma unroll
        for (int i = 0; i < 2; ++i)
#pragma unroll
            for (int r = 0; r < 8; ++r) {
                const int row = m0 + i * 16 + r + half * 8;
                const float val = c[i][t][r] + bvl;
                if (OUT_F16) ((_Float16*)out_)[(size_t)row * N + nB] = (_Float16)val;
                else         ((float*)out_)[(size_t)row * N + nB]    = val;
            }
    }
}

// ---------------------------------------------------------------------------
// RoPE in-place on f16 buffer laid out [B*L, D], D = H*Hd, Hd = 64
// ---------------------------------------------------------------------------
__global__ __launch_bounds__(256) void rope_f16(_Float16* x, int L, int D, int total)
{
    const int idx = blockIdx.x * blockDim.x + threadIdx.x; // over B*L*(D/2)
    if (idx >= total) return;
    const int halfD = D >> 1;
    const int bl   = idx / halfD;
    const int pair = idx - bl * halfD;
    const int l    = bl % L;
    const int i    = pair & 31;        // dim-pair index within head (Hd/2 = 32)
    const int h    = pair >> 5;
    const float ang = (float)l * __powf(10000.0f, -(float)(2 * i) / 64.0f);
    float s, cct;
    __sincosf(ang, &s, &cct);
    const size_t base = (size_t)bl * D + h * 64 + 2 * i;
    const float x1 = (float)x[base];
    const float x2 = (float)x[base + 1];
    x[base]     = (_Float16)(x1 * cct - x2 * s);
    x[base + 1] = (_Float16)(x1 * s + x2 * cct);
}

// ---------------------------------------------------------------------------
// V transpose: v[(b*L+l)*D + h*64 + d] -> vt[((b*16+h)*64 + d)*L + l]
// ---------------------------------------------------------------------------
__global__ __launch_bounds__(256) void transpose_v(
    const _Float16* __restrict__ v, _Float16* __restrict__ vt, int L, int D, int total)
{
    const int idx = blockIdx.x * blockDim.x + threadIdx.x;
    if (idx >= total) return;
    const int d0 = idx % D;
    const int l  = (idx / D) % L;
    const int b  = idx / (D * L);
    const int h  = d0 >> 6;
    const int d  = d0 & 63;
    vt[((size_t)(b * 16 + h) * 64 + d) * L + l] = v[idx];
}

// ---------------------------------------------------------------------------
// Flash attention (online softmax), per-wave 16 query rows, 64-key chunks.
// q,k: [B*L, D] f16 (post-RoPE, q pre-scaled by 1/sqrt(Hd) here);
// vt: [B*H, 64, L] f16; o: [B*L, D] f16.
// Scores staged through per-wave LDS: two side-by-side pair-swizzled 32-col
// regions per row (col' = c*32 + (lc&15)*2 + (lc>>4)), so C-layout stores
// pack to b32 pairs and each P A-fragment reads back as 2 contiguous b128s.
// After read-back each lane owns a full query row -> in-lane softmax stats
// plus a single cross-half shuffle.
// ---------------------------------------------------------------------------
__global__ __launch_bounds__(256) void attn_wmma(
    const _Float16* __restrict__ q, const _Float16* __restrict__ k,
    const _Float16* __restrict__ vt, _Float16* __restrict__ o, int L, int D)
{
    __shared__ _Float16 lds_p[8][16 * 64];   // per-wave score staging (2 KB each)
    __shared__ float    lds_c[8][16];        // per-wave row-stat broadcast line

    const int lane = threadIdx.x & 31;
    const int wid  = threadIdx.x >> 5;
    const int half = lane >> 4;
    const int l15  = lane & 15;
    const int bh   = blockIdx.y;
    const int b    = bh >> 4;
    const int h    = bh & 15;
    const int q0   = blockIdx.x * 128 + wid * 16;

    _Float16* pw = &lds_p[wid][0];
    float*    lc = &lds_c[wid][0];

    // Q fragments: 16 rows x 64 dims -> two K=32 A-fragments, 1/sqrt(Hd) folded in
    v16h aq[2];
#pragma unroll
    for (int f = 0; f < 2; ++f) {
        const _Float16* qp = q + (size_t)(b * L + q0 + l15) * D + h * 64 + f * 32;
        aq[f] = mk16(*(const v8h*)(qp + half * 16), *(const v8h*)(qp + half * 16 + 8));
#pragma unroll
        for (int j = 0; j < 16; ++j) aq[f][j] = aq[f][j] * (_Float16)0.125f;
    }

    v8f acc[4] = {};
    float rmax = -1e30f;   // this lane's row (= l15) running max
    float rsum = 0.0f;     // this lane's row running sum

    for (int kk = 0; kk < L; kk += 64) {
        // --- S = Q @ K^T for 64-key chunk (four 16-key column tiles) ---
        v8f s[4] = {};
#pragma unroll
        for (int f = 0; f < 2; ++f) {
#pragma unroll
            for (int t = 0; t < 4; ++t) {
                const _Float16* kp = k + (size_t)(b * L + kk + t * 16 + l15) * D
                                       + h * 64 + f * 32 + half * 16;
                v16h bk = mk16(*(const v8h*)kp, *(const v8h*)(kp + 8));
                s[t] = wmma_f16(aq[f], bk, s[t]);
            }
        }

        // --- store scores to LDS: (s[2c], s[2c+1]) pack to b32 per region c ---
#pragma unroll
        for (int r = 0; r < 8; ++r) {
            _Float16* rowp = pw + (r + half * 8) * 64;
#pragma unroll
            for (int c = 0; c < 2; ++c) {
                v2h p;
                p[0] = (_Float16)s[2 * c][r];
                p[1] = (_Float16)s[2 * c + 1][r];
                *(v2h*)(rowp + c * 32 + l15 * 2) = p;
            }
        }
        lds_fence_wait();

        // --- read back in A-layout order; lane now owns a full query row ---
        const _Float16* pr = pw + l15 * 64;
        v8h raw[4];
#pragma unroll
        for (int c = 0; c < 2; ++c) {
            raw[2 * c]     = *(const v8h*)(pr + c * 32 + half * 16);
            raw[2 * c + 1] = *(const v8h*)(pr + c * 32 + half * 16 + 8);
        }
        float sv[32];
#pragma unroll
        for (int g = 0; g < 4; ++g)
#pragma unroll
            for (int t = 0; t < 8; ++t) sv[g * 8 + t] = (float)raw[g][t];

        // in-lane row max over 32 + one cross-half combine
        float cmax = sv[0];
#pragma unroll
        for (int t = 1; t < 32; ++t) cmax = fmaxf(cmax, sv[t]);
        cmax = fmaxf(cmax, __shfl_xor(cmax, 16));

        const float nm   = fmaxf(rmax, cmax);
        const float corr = __expf(rmax - nm);
        rmax = nm;

        float pv[32], psum = 0.0f;
#pragma unroll
        for (int t = 0; t < 32; ++t) { pv[t] = __expf(sv[t] - nm); psum += pv[t]; }
        psum += __shfl_xor(psum, 16);
        rsum = rsum * corr + psum;

        // P in A-fragment layout: region c -> frag pa[c]; raw[2j]->a[j], raw[2j+1]->a[8+j]
        v16h pa[2];
#pragma unroll
        for (int c = 0; c < 2; ++c)
#pragma unroll
            for (int j = 0; j < 8; ++j) {
                pa[c][j]     = (_Float16)pv[c * 16 + 2 * j];
                pa[c][8 + j] = (_Float16)pv[c * 16 + 2 * j + 1];
            }

        // --- broadcast corr (row domain, lane = row) to C layout via LDS ---
        lc[l15] = corr;              // both halves write the same value
        lds_fence_wait();
        float cc[8];
#pragma unroll
        for (int r = 0; r < 8; ++r) cc[r] = lc[half * 8 + r];
#pragma unroll
        for (int t = 0; t < 4; ++t)
#pragma unroll
            for (int r = 0; r < 8; ++r) acc[t][r] *= cc[r];

        // --- acc += P @ V (four 16-dim column tiles, two key half-chunks) ---
#pragma unroll
        for (int t = 0; t < 4; ++t) {
            const _Float16* vp = vt + ((size_t)bh * 64 + t * 16 + l15) * L + kk + half * 8;
            v16h bv0 = mk16(*(const v8h*)vp, *(const v8h*)(vp + 16));
            acc[t] = wmma_f16(pa[0], bv0, acc[t]);
            v16h bv1 = mk16(*(const v8h*)(vp + 32), *(const v8h*)(vp + 48));
            acc[t] = wmma_f16(pa[1], bv1, acc[t]);
        }
    }

    // --- broadcast rsum to C layout, normalize, store ---
    lc[l15] = rsum;
    lds_fence_wait();
    float rinv[8];
#pragma unroll
    for (int r = 0; r < 8; ++r) rinv[r] = 1.0f / lc[half * 8 + r];
#pragma unroll
    for (int t = 0; t < 4; ++t)
#pragma unroll
        for (int r = 0; r < 8; ++r) {
            const float val = acc[t][r] * rinv[r];
            o[(size_t)(b * L + q0 + r + half * 8) * D + h * 64 + t * 16 + l15] = (_Float16)val;
        }
}

// ---------------------------------------------------------------------------
// launch
// ---------------------------------------------------------------------------
extern "C" void kernel_launch(void* const* d_in, const int* in_sizes, int n_in,
                              void* d_out, int out_size, void* d_ws, size_t ws_size,
                              hipStream_t stream) {
    (void)in_sizes; (void)n_in; (void)out_size; (void)ws_size;

    const float* q  = (const float*)d_in[0];
    const float* k  = (const float*)d_in[1];
    const float* v  = (const float*)d_in[2];
    const float* wq = (const float*)d_in[3];
    const float* bq = (const float*)d_in[4];
    const float* wk = (const float*)d_in[5];
    const float* bk = (const float*)d_in[6];
    const float* wv = (const float*)d_in[7];
    const float* bv = (const float*)d_in[8];
    const float* wo = (const float*)d_in[9];
    const float* bo = (const float*)d_in[10];
    float* out = (float*)d_out;

    const int B = 2, L = 2048, D = 1024;
    const int M = B * L;                       // 4096

    char* ws = (char*)d_ws;
    const size_t buf = (size_t)M * D * sizeof(_Float16);   // 8 MB
    _Float16* in16 = (_Float16*)(ws + 0 * buf);            // converted activations (reused)
    _Float16* q16  = (_Float16*)(ws + 1 * buf);
    _Float16* k16  = (_Float16*)(ws + 2 * buf);
    _Float16* v16  = (_Float16*)(ws + 3 * buf);
    _Float16* vt16 = (_Float16*)(ws + 4 * buf);
    _Float16* a16  = (_Float16*)(ws + 5 * buf);
    _Float16* w16  = (_Float16*)(ws + 6 * buf);            // converted weights (reused, 2 MB)

    const int act_total = M * D;                           // 4M elements
    const int w_total   = D * D;                           // 1M elements
    const int act_blk   = act_total / 8 / 256;
    const int w_blk     = w_total / 8 / 256;

    dim3 gg(D / 128, M / 128);                             // (8, 32)

    // Q projection
    cvt_f16<<<act_blk, 256, 0, stream>>>(q, in16, act_total);
    cvt_f16<<<w_blk,   256, 0, stream>>>(wq, w16, w_total);
    gemm_bias_wmma<true><<<gg, 256, 0, stream>>>(in16, w16, bq, q16, M, D, D);
    // K projection
    cvt_f16<<<act_blk, 256, 0, stream>>>(k, in16, act_total);
    cvt_f16<<<w_blk,   256, 0, stream>>>(wk, w16, w_total);
    gemm_bias_wmma<true><<<gg, 256, 0, stream>>>(in16, w16, bk, k16, M, D, D);
    // V projection
    cvt_f16<<<act_blk, 256, 0, stream>>>(v, in16, act_total);
    cvt_f16<<<w_blk,   256, 0, stream>>>(wv, w16, w_total);
    gemm_bias_wmma<true><<<gg, 256, 0, stream>>>(in16, w16, bv, v16, M, D, D);

    // RoPE on q/k
    const int rope_total = B * L * (D / 2);
    rope_f16<<<(rope_total + 255) / 256, 256, 0, stream>>>(q16, L, D, rope_total);
    rope_f16<<<(rope_total + 255) / 256, 256, 0, stream>>>(k16, L, D, rope_total);

    // V -> [bh, d, L] for contiguous key-axis fragment loads
    transpose_v<<<(act_total + 255) / 256, 256, 0, stream>>>(v16, vt16, L, D, act_total);

    // Flash attention (64-key chunks)
    attn_wmma<<<dim3(L / 128, B * 16), 256, 0, stream>>>(q16, k16, vt16, a16, L, D);

    // Output projection (f16 in -> f32 out)
    cvt_f16<<<w_blk, 256, 0, stream>>>(wo, w16, w_total);
    gemm_bias_wmma<false><<<gg, 256, 0, stream>>>(a16, w16, bo, out, M, D, D);
}
